// MultiHeadAttentionQuantum_65481071409502
// MI455X (gfx1250) — compile-verified, matrix operand order checked by reading, not run on hardware
//
#include <hip/hip_runtime.h>
#include <hip/hip_bf16.h>
#include <stdint.h>

// MI455X (gfx1250) wave32 WMMA + Tensor-Data-Mover implementation of
// quantum-MHA. bf16 operands + f32 accumulation (v_wmma_f32_16x16x32_bf16);
// flash attention keeps the 512MB score tensor out of HBM; K/V tiles are
// double-buffered through LDS via TDM (tensor_load_to_lds / TENSORcnt).

#define E_DIM   1024
#define NHEADS  16
#define DK      64
#define SEQ     2048
#define BATCH   2
#define NROWS   (BATCH * SEQ)   // 4096

typedef __bf16 bf16_t;
typedef __attribute__((ext_vector_type(16))) __bf16 bf16x16;
typedef __attribute__((ext_vector_type(8)))  float  f32x8;
typedef __attribute__((ext_vector_type(4)))  unsigned int u32x4;
typedef __attribute__((ext_vector_type(8)))  int          i32x8;
typedef __attribute__((ext_vector_type(4)))  int          i32x4;

#if __has_builtin(__builtin_amdgcn_tensor_load_to_lds) && \
    __has_builtin(__builtin_amdgcn_s_wait_tensorcnt)
#define HAVE_TDM 1
#if __has_include(<hip/amd_detail/amd_gfx1250_TDM.h>)
#define TDM_6ARG 1
#endif
#endif

struct Pack32 { uint4 lo, hi; };

static __device__ __forceinline__ bf16x16 make_frag(uint4 lo, uint4 hi) {
  Pack32 p{lo, hi};
  return __builtin_bit_cast(bf16x16, p);
}

static __device__ __forceinline__ unsigned short f2bf(float f) {
  bf16_t h = (bf16_t)f;              // native v_cvt path, RNE
  return __builtin_bit_cast(unsigned short, h);
}

static __device__ __forceinline__ unsigned pack2bf(float a, float b) {
  return (unsigned)f2bf(a) | ((unsigned)f2bf(b) << 16);
}

// 16x32 bf16 fragment (A or B role), contiguous-K source. Per ISA 16-bit
// layout: lane = (M or N) & 15, half = lane>>4 selects the K interleave.
// Two 16B chunks: K in [kOff+half*8,+8) and [kOff+16+half*8,+8).
static __device__ __forceinline__ bf16x16
ld_frag_lds(const unsigned short* base, int row0, int stride, int kOff, int lane) {
  const int m = lane & 15, half = lane >> 4;
  const unsigned short* p = base + (size_t)(row0 + m) * stride + kOff + half * 8;
  uint4 lo = *(const uint4*)p;
  uint4 hi = *(const uint4*)(p + 16);
  return make_frag(lo, hi);
}

static __device__ __forceinline__ bf16x16
ld_frag_global(const bf16_t* base, int row0, int stride, int kOff, int lane) {
  const int m = lane & 15, half = lane >> 4;
  const bf16_t* p = base + (size_t)(row0 + m) * stride + kOff + half * 8;
  uint4 lo = *(const uint4*)p;
  uint4 hi = *(const uint4*)(p + 16);
  return make_frag(lo, hi);
}

// B fragment where the K dimension runs over LDS *rows* (strided gather):
// K = jOff + half*8 + e (e<8), jOff + 16 + half*8 + (e-8) (e>=8);
// lane supplies column n = lane&15 at 'col0 + n'.
static __device__ __forceinline__ bf16x16
ld_frag_b_rows(const unsigned short* base, int jOff, int stride, int col0, int lane) {
  const int n = lane & 15, half = lane >> 4;
  const unsigned short* p = base + (size_t)(jOff + half * 8) * stride + col0 + n;
  struct S16 { unsigned short a[16]; } s;
#pragma unroll
  for (int e = 0; e < 8; ++e) s.a[e] = p[e * stride];
  const unsigned short* q = p + 16 * stride;
#pragma unroll
  for (int e = 0; e < 8; ++e) s.a[8 + e] = q[e * stride];
  return __builtin_bit_cast(bf16x16, s);
}

static __device__ __forceinline__ f32x8 wmma_bf16(bf16x16 a, bf16x16 b, f32x8 c) {
  return __builtin_amdgcn_wmma_f32_16x16x32_bf16(false, a, false, b, (short)0, c,
                                                 false, false);
}

#ifdef HAVE_TDM
// TDM: load a tileH x tileW bf16 tile (rows contiguous, rowStride elements in
// memory) into LDS at ldsByteOff, padding each LDS row per pad codes so the
// LDS row stride matches the fragment loaders. D# per ISA ch.8.
static __device__ __forceinline__ void tdm_load_2d_bf16(
    unsigned ldsByteOff, const void* gsrc,
    unsigned tileW, unsigned tileH, unsigned long long rowStrideElems,
    unsigned padIntervalCode, unsigned padAmountCode) {
  const unsigned long long ga = (unsigned long long)(size_t)gsrc;
  u32x4 g0;
  g0[0] = 1u;                                           // count=1, no gather
  g0[1] = ldsByteOff;                                   // lds_addr (bytes)
  g0[2] = (unsigned)(ga & 0xffffffffu);                 // global_addr lo
  g0[3] = (unsigned)((ga >> 32) & 0x01ffffffu) | (2u << 30);  // addr hi | type=2
  i32x8 g1;
  g1[0] = (int)((1u << 16) | (1u << 20) |               // data_size=2B, pad_en
                (padIntervalCode << 22) | (padAmountCode << 25));
  g1[1] = (int)(tileW << 16);                           // tensor_dim0 lo16
  g1[2] = (int)(tileH << 16);                           // dim0 hi | tensor_dim1 lo16
  g1[3] = (int)(tileW << 16);                           // dim1 hi | tile_dim0
  g1[4] = (int)tileH;                                   // tile_dim1 | tile_dim2=0
  g1[5] = (int)(unsigned)(rowStrideElems & 0xffffffffu);// dim0_stride lo32
  g1[6] = (int)(unsigned)((rowStrideElems >> 32) & 0xffffu); // hi16 | dim1_str lo
  g1[7] = 0;
  const i32x4 z4 = {0, 0, 0, 0};
#ifdef TDM_6ARG
  const i32x8 z8 = {0, 0, 0, 0, 0, 0, 0, 0};
  __builtin_amdgcn_tensor_load_to_lds(g0, g1, z4, z4, z8, 0);
#else
  __builtin_amdgcn_tensor_load_to_lds(g0, g1, z4, z4, 0);
#endif
}
#endif

// ---------------------------------------------------------------------------
// Kernel A: Y = X @ W^T per 128x128 tile; epilogue cos(y + theta[col%64]),
// scatter to head-major bf16 [B,H,S,DK]. blockIdx.z selects Wq/Wk/Wv.
// (f32 inputs must be converted to bf16 anyway, so staging stays manual.)
// ---------------------------------------------------------------------------
__global__ __launch_bounds__(256) void qkv_proj_kernel(
    const float* __restrict__ x,
    const float* __restrict__ Wq, const float* __restrict__ Wk,
    const float* __restrict__ Wv, const float* __restrict__ theta,
    bf16_t* __restrict__ qh, bf16_t* __restrict__ kh, bf16_t* __restrict__ vh) {
  __shared__ unsigned short Xs[128 * 40];
  __shared__ unsigned short Ws[128 * 40];

  const int tid = threadIdx.x;
  const int lane = tid & 31;
  const int wid = tid >> 5;
  const int rowBase = blockIdx.x * 128;
  const int colBase = blockIdx.y * 128;
  const float* W  = (blockIdx.z == 0) ? Wq : (blockIdx.z == 1) ? Wk : Wv;
  bf16_t*     dst = (blockIdx.z == 0) ? qh : (blockIdx.z == 1) ? kh : vh;

  const int waveRow = (wid & 3) * 32;
  const int waveCol = (wid >> 2) * 64;

  const f32x8 zero = {0.f, 0.f, 0.f, 0.f, 0.f, 0.f, 0.f, 0.f};
  f32x8 acc[2][4];
  for (int i = 0; i < 2; ++i)
    for (int j = 0; j < 4; ++j) acc[i][j] = zero;

  for (int kt = 0; kt < E_DIM; kt += 32) {
    __syncthreads();
    for (int it = 0; it < 4; ++it) {
      int f = tid + it * 256;     // float4 index, 0..1023
      int r = f >> 3;
      int c = (f & 7) * 4;
      float4 xv = *(const float4*)(x + (size_t)(rowBase + r) * E_DIM + kt + c);
      float4 wv = *(const float4*)(W + (size_t)(colBase + r) * E_DIM + kt + c);
      *(uint2*)(&Xs[r * 40 + c]) = make_uint2(pack2bf(xv.x, xv.y), pack2bf(xv.z, xv.w));
      *(uint2*)(&Ws[r * 40 + c]) = make_uint2(pack2bf(wv.x, wv.y), pack2bf(wv.z, wv.w));
    }
    __syncthreads();

    bf16x16 a0 = ld_frag_lds(Xs, waveRow,      40, 0, lane);
    bf16x16 a1 = ld_frag_lds(Xs, waveRow + 16, 40, 0, lane);
#pragma unroll
    for (int ni = 0; ni < 4; ++ni) {
      bf16x16 bfr = ld_frag_lds(Ws, waveCol + ni * 16, 40, 0, lane);
      acc[0][ni] = wmma_bf16(a0, bfr, acc[0][ni]);
      acc[1][ni] = wmma_bf16(a1, bfr, acc[1][ni]);
    }
  }

  const int half = lane >> 4, nlo = lane & 15;
#pragma unroll
  for (int mi = 0; mi < 2; ++mi) {
#pragma unroll
    for (int ni = 0; ni < 4; ++ni) {
      const int col = colBase + waveCol + ni * 16 + nlo;
      const int h = col >> 6;
      const int d = col & 63;
      const float th = theta[d];
#pragma unroll
      for (int r = 0; r < 8; ++r) {
        const int row = rowBase + waveRow + mi * 16 + r + half * 8;
        const int b = row >> 11;
        const int s = row & (SEQ - 1);
        const float val = __cosf(acc[mi][ni][r] + th);
        const size_t off = (((size_t)(b * NHEADS + h) * SEQ) + s) * DK + d;
        ((unsigned short*)dst)[off] = f2bf(val);
      }
    }
  }
}

// ---------------------------------------------------------------------------
// Kernel B: flash attention per (b,h). WG = 128 Q rows; wave = 16 rows.
// K/V blocks (64 rows x 64 d) double-buffered in LDS via TDM; online softmax
// in f32 with 16-lane shfl reductions; P round-trips through per-wave LDS.
// ---------------------------------------------------------------------------
__global__ __launch_bounds__(256) void flash_attn_kernel(
    const bf16_t* __restrict__ qh, const bf16_t* __restrict__ kh,
    const bf16_t* __restrict__ vh, bf16_t* __restrict__ oh) {
  __shared__ unsigned short Ks[2][64 * 72];   // [j][d], padded rows (144B)
  __shared__ unsigned short Vs[2][64 * 72];   // [j][d]
  __shared__ unsigned short Ps[8][16 * 72];   // per-wave P staging [i][j]

  const int tid = threadIdx.x;
  const int lane = tid & 31;
  const int wid = tid >> 5;
  const int bh = blockIdx.y;
  const int qBlock = blockIdx.x * 128;
  const int b = bh >> 4;
  const int h = bh & 15;

  const bf16_t* qBase = qh + (size_t)bh * SEQ * DK;
  const bf16_t* kBase = kh + (size_t)bh * SEQ * DK;
  const bf16_t* vBase = vh + (size_t)bh * SEQ * DK;

  const int waveRow = wid * 16;
  const int half = lane >> 4, nlo = lane & 15;

  bf16x16 qf0 = ld_frag_global(qBase, qBlock + waveRow, DK, 0,  lane);
  bf16x16 qf1 = ld_frag_global(qBase, qBlock + waveRow, DK, 32, lane);

  const f32x8 zero = {0.f, 0.f, 0.f, 0.f, 0.f, 0.f, 0.f, 0.f};
  f32x8 oAcc[4];
  for (int i = 0; i < 4; ++i) oAcc[i] = zero;
  float mrun[8], lrun[8];
#pragma unroll
  for (int r = 0; r < 8; ++r) { mrun[r] = -3.0e38f; lrun[r] = 0.f; }

  const int nBlk = SEQ / 64;
#ifdef HAVE_TDM
  // preload block 0 into buffer 0 (DMA; pad 32 DWORDs -> +4 DWORDs = 144B rows)
  if (wid == 0) {
    tdm_load_2d_bf16((unsigned)(size_t)&Ks[0][0], kBase, 64, 64, 64, 4, 3);
    tdm_load_2d_bf16((unsigned)(size_t)&Vs[0][0], vBase, 64, 64, 64, 4, 3);
  }
#endif

  for (int i = 0; i < nBlk; ++i) {
    const int buf = i & 1;
    const unsigned short* ks = Ks[buf];
    const unsigned short* vs = Vs[buf];

#ifdef HAVE_TDM
    if (wid == 0) __builtin_amdgcn_s_wait_tensorcnt(0);  // current buffer ready
    __syncthreads();   // publish; also retires prior reads of the other buffer
    if (wid == 0 && i + 1 < nBlk) {
      const size_t nxt = (size_t)(i + 1) * 64 * DK;
      tdm_load_2d_bf16((unsigned)(size_t)&Ks[buf ^ 1][0], kBase + nxt, 64, 64, 64, 4, 3);
      tdm_load_2d_bf16((unsigned)(size_t)&Vs[buf ^ 1][0], vBase + nxt, 64, 64, 64, 4, 3);
    }
#else
    if (i + 1 < nBlk) {
      __builtin_prefetch(kBase + (size_t)(i + 1) * 64 * DK + (size_t)(tid >> 2) * DK, 0, 1);
      __builtin_prefetch(vBase + (size_t)(i + 1) * 64 * DK + (size_t)(tid >> 2) * DK, 0, 1);
    }
    __syncthreads();
    for (int it = 0; it < 2; ++it) {
      int f = tid + it * 256;   // uint4 index, 8 per 64-elem row
      int j = f >> 3;
      int c = (f & 7) * 8;
      *(uint4*)(&Ks[buf][j * 72 + c]) = *(const uint4*)(kBase + ((size_t)i * 64 + j) * DK + c);
      *(uint4*)(&Vs[buf][j * 72 + c]) = *(const uint4*)(vBase + ((size_t)i * 64 + j) * DK + c);
    }
    __syncthreads();
#endif

    // S tile: 16 rows x 64 cols, K=64 in two WMMA steps per 16-col tile
    f32x8 sfr[4];
#pragma unroll
    for (int jt = 0; jt < 4; ++jt) {
      bf16x16 kf0 = ld_frag_lds(ks, jt * 16, 72, 0,  lane);
      bf16x16 kf1 = ld_frag_lds(ks, jt * 16, 72, 32, lane);
      f32x8 sc = zero;
      sc = wmma_bf16(qf0, kf0, sc);
      sc = wmma_bf16(qf1, kf1, sc);
      sfr[jt] = sc;
    }

    // online softmax (scale 1/sqrt(64) = 0.125)
    float mnew[8];
#pragma unroll
    for (int r = 0; r < 8; ++r) mnew[r] = mrun[r];
#pragma unroll
    for (int jt = 0; jt < 4; ++jt)
#pragma unroll
      for (int r = 0; r < 8; ++r) {
        float v = sfr[jt][r] * 0.125f;
        sfr[jt][r] = v;
        mnew[r] = fmaxf(mnew[r], v);
      }
#pragma unroll
    for (int r = 0; r < 8; ++r) {
      float v = mnew[r];
      for (int off = 8; off >= 1; off >>= 1) v = fmaxf(v, __shfl_xor(v, off, 32));
      mnew[r] = v;
    }
    float alpha[8], rsum[8];
#pragma unroll
    for (int r = 0; r < 8; ++r) { alpha[r] = __expf(mrun[r] - mnew[r]); rsum[r] = 0.f; }

    unsigned short* myP = Ps[wid];
#pragma unroll
    for (int jt = 0; jt < 4; ++jt)
#pragma unroll
      for (int r = 0; r < 8; ++r) {
        float p = __expf(sfr[jt][r] - mnew[r]);
        rsum[r] += p;
        myP[(r + half * 8) * 72 + jt * 16 + nlo] = f2bf(p);
      }
#pragma unroll
    for (int r = 0; r < 8; ++r) {
      float v = rsum[r];
      for (int off = 8; off >= 1; off >>= 1) v += __shfl_xor(v, off, 32);
      lrun[r] = lrun[r] * alpha[r] + v;
      mrun[r] = mnew[r];
#pragma unroll
      for (int dt = 0; dt < 4; ++dt) oAcc[dt][r] *= alpha[r];
    }

    // O += P(16x64) @ V(64x64). Per-wave LDS ops are in-order, so the P
    // read-after-write needs no barrier; V B-frags gathered across rows.
#pragma unroll
    for (int kt = 0; kt < 2; ++kt) {
      bf16x16 pf = ld_frag_lds(myP, 0, 72, kt * 32, lane);
#pragma unroll
      for (int dt = 0; dt < 4; ++dt) {
        bf16x16 vf = ld_frag_b_rows(vs, kt * 32, 72, dt * 16, lane);
        oAcc[dt] = wmma_bf16(pf, vf, oAcc[dt]);
      }
    }
  }

#pragma unroll
  for (int dt = 0; dt < 4; ++dt)
#pragma unroll
    for (int r = 0; r < 8; ++r) {
      const int s = qBlock + waveRow + r + half * 8;
      const float val = oAcc[dt][r] / lrun[r];
      const size_t off = ((size_t)(b * SEQ + s)) * E_DIM + h * 64 + dt * 16 + nlo;
      ((unsigned short*)oh)[off] = f2bf(val);
    }
}

// ---------------------------------------------------------------------------
// Kernel C: out = OH(bf16) @ Wo^T, f32 output. OH tile fetched by TDM
// (already bf16) while Wo is converted f32->bf16 by the VALU concurrently.
// ---------------------------------------------------------------------------
__global__ __launch_bounds__(256) void out_proj_kernel(
    const bf16_t* __restrict__ oh, const float* __restrict__ Wo,
    float* __restrict__ out) {
  __shared__ unsigned short Xs[128 * 40];
  __shared__ unsigned short Ws[128 * 40];

  const int tid = threadIdx.x;
  const int lane = tid & 31;
  const int wid = tid >> 5;
  const int rowBase = blockIdx.x * 128;
  const int colBase = blockIdx.y * 128;
  const int waveRow = (wid & 3) * 32;
  const int waveCol = (wid >> 2) * 64;

  const f32x8 zero = {0.f, 0.f, 0.f, 0.f, 0.f, 0.f, 0.f, 0.f};
  f32x8 acc[2][4];
  for (int i = 0; i < 2; ++i)
    for (int j = 0; j < 4; ++j) acc[i][j] = zero;

  for (int kt = 0; kt < E_DIM; kt += 32) {
    __syncthreads();
#ifdef HAVE_TDM
    // 128 rows x 32 bf16; row = 16 DWORDs, pad +4 DWORDs -> 80B LDS rows
    if (wid == 0)
      tdm_load_2d_bf16((unsigned)(size_t)&Xs[0],
                       oh + (size_t)rowBase * E_DIM + kt, 32, 128, E_DIM, 3, 3);
#else
    for (int it = 0; it < 2; ++it) {
      int f = tid + it * 256;
      int r = f >> 2;
      int c = (f & 3) * 8;
      *(uint4*)(&Xs[r * 40 + c]) = *(const uint4*)(oh + (size_t)(rowBase + r) * E_DIM + kt + c);
    }
#endif
    for (int it = 0; it < 4; ++it) {
      int f = tid + it * 256;
      int r = f >> 3;
      int c = (f & 7) * 4;
      float4 wv = *(const float4*)(Wo + (size_t)(colBase + r) * E_DIM + kt + c);
      *(uint2*)(&Ws[r * 40 + c]) = make_uint2(pack2bf(wv.x, wv.y), pack2bf(wv.z, wv.w));
    }
#ifdef HAVE_TDM
    if (wid == 0) __builtin_amdgcn_s_wait_tensorcnt(0);
#endif
    __syncthreads();

    bf16x16 a0 = ld_frag_lds(Xs, waveRow,      40, 0, lane);
    bf16x16 a1 = ld_frag_lds(Xs, waveRow + 16, 40, 0, lane);
#pragma unroll
    for (int ni = 0; ni < 4; ++ni) {
      bf16x16 bfr = ld_frag_lds(Ws, waveCol + ni * 16, 40, 0, lane);
      acc[0][ni] = wmma_bf16(a0, bfr, acc[0][ni]);
      acc[1][ni] = wmma_bf16(a1, bfr, acc[1][ni]);
    }
  }

  const int half = lane >> 4, nlo = lane & 15;
#pragma unroll
  for (int mi = 0; mi < 2; ++mi)
#pragma unroll
    for (int ni = 0; ni < 4; ++ni) {
      const int col = colBase + waveCol + ni * 16 + nlo;
#pragma unroll
      for (int r = 0; r < 8; ++r) {
        const int row = rowBase + waveRow + mi * 16 + r + half * 8;
        out[(size_t)row * E_DIM + col] = acc[mi][ni][r];
      }
    }
}

// ---------------------------------------------------------------------------
extern "C" void kernel_launch(void* const* d_in, const int* in_sizes, int n_in,
                              void* d_out, int out_size, void* d_ws, size_t ws_size,
                              hipStream_t stream) {
  const float* x     = (const float*)d_in[0];
  const float* Wq    = (const float*)d_in[1];
  const float* Wk    = (const float*)d_in[2];
  const float* Wv    = (const float*)d_in[3];
  const float* Wo    = (const float*)d_in[4];
  const float* theta = (const float*)d_in[5];
  float* out = (float*)d_out;

  const size_t HSZ = (size_t)BATCH * NHEADS * SEQ * DK;  // 4M elements
  bf16_t* qh = (bf16_t*)d_ws;
  bf16_t* kh = qh + HSZ;
  bf16_t* vh = kh + HSZ;
  bf16_t* oh = vh + HSZ;   // merged-head [B,S,E] bf16

  dim3 gA(NROWS / 128, E_DIM / 128, 3);
  qkv_proj_kernel<<<gA, 256, 0, stream>>>(x, Wq, Wk, Wv, theta, qh, kh, vh);

  dim3 gB(SEQ / 128, BATCH * NHEADS);
  flash_attn_kernel<<<gB, 256, 0, stream>>>(qh, kh, vh, oh);

  dim3 gC(NROWS / 128, E_DIM / 128);
  out_proj_kernel<<<gC, 256, 0, stream>>>(oh, Wo, out);
}
